// BiLSTM_geeky_30399778521235
// MI455X (gfx1250) — matrix operand order repaired and checked by reading, not compiled, not run on hardware
//
#include <hip/hip_runtime.h>
#include <hip/hip_bf16.h>
#include <stdint.h>

typedef __attribute__((ext_vector_type(16))) _Float16 v16h;
typedef __attribute__((ext_vector_type(8)))  float    v8f;

// Problem constants (match reference)
constexpr int I_ = 128;    // input size
constexpr int H_ = 256;    // hidden size
constexpr int B_ = 256;    // batch
constexpr int T_ = 512;    // time steps
constexpr int G_ = 4 * H_; // 1024 gate columns (i,f,g,o)

constexpr int NT  = G_ / 16;  // 64 N-tiles of 16 columns
constexpr int KTH = H_ / 32;  // 8 K-tiles (recurrent GEMM)
constexpr int KTX = I_ / 32;  // 4 K-tiles (input GEMM)

constexpr int WHH_FRAG_DW = NT * KTH * 32 * 8; // 131072 dwords per direction
constexpr int WIH_FRAG_DW = NT * KTX * 32 * 8; //  65536 dwords per direction

union Frag16 { v16h v; uint4 q[2]; };

// ---------------------------------------------------------------------------
// Pack W (f32, [G rows = gate index n][K cols = k]) into WMMA B-fragment order:
// dword index = ((nt*KT + kt)*32 + lane)*8 + v
// B element (k, n): n = nt*16 + lane%16 ; k = kt*32 + (lane/16)*16 + 2v (+1)
// ---------------------------------------------------------------------------
__global__ void conv_b_frag_kernel(const float* __restrict__ W,
                                   unsigned int* __restrict__ out,
                                   int K, int KT) {
  int idx = blockIdx.x * blockDim.x + threadIdx.x;
  int total = NT * KT * 32 * 8;
  if (idx >= total) return;
  int v    = idx & 7;
  int lane = (idx >> 3) & 31;
  int kt   = (idx >> 8) % KT;
  int nt   = (idx >> 8) / KT;
  int n = nt * 16 + (lane & 15);
  int k = kt * 32 + (lane >> 4) * 16 + 2 * v;
  union { unsigned int u; _Float16 h[2]; } p;
  p.h[0] = (_Float16)W[(size_t)n * K + k];
  p.h[1] = (_Float16)W[(size_t)n * K + k + 1];
  out[idx] = p.u;
}

__global__ void bias_kernel(const float* __restrict__ bih_f, const float* __restrict__ bhh_f,
                            const float* __restrict__ bih_b, const float* __restrict__ bhh_b,
                            float* __restrict__ biasC) {
  int idx = blockIdx.x * blockDim.x + threadIdx.x;
  if (idx >= 2 * G_) return;
  int dir = idx >> 10, n = idx & (G_ - 1);
  biasC[idx] = dir ? (bih_b[n] + bhh_b[n]) : (bih_f[n] + bhh_f[n]);
}

// ---------------------------------------------------------------------------
// Persistent bidirectional-LSTM kernel.
// grid = (16 batch-chunks, 2 directions), block = 256 threads (8 wave32).
// Each WG owns 16 batch rows; recurrence is independent per batch row, so no
// cross-WG sync. Wave w computes N-tiles [8w, 8w+8) of the 64 gate tiles.
// ---------------------------------------------------------------------------
__global__ __launch_bounds__(256, 1)
void lstm_kernel(const float* __restrict__ x,
                 const unsigned int* __restrict__ whhF,
                 const unsigned int* __restrict__ wihF,
                 const float* __restrict__ biasC,
                 float* __restrict__ hfinal) {
  __shared__ _Float16 h_sh[16][H_];    //  8 KB  current h (A-matrix source)
  __shared__ _Float16 xa_sh[16][I_];   //  4 KB  x_t tile in f16
  __shared__ float    gates_sh[G_][16];// 64 KB  gate pre-activations [n][m]
  __shared__ float    c_sh[H_][16];    // 16 KB  cell state [j][m]
  __shared__ float    h32_sh[H_][16];  // 16 KB  h in f32 [j][m]

  const int tid   = threadIdx.x;
  const int lane  = tid & 31;
  const int wave  = tid >> 5;          // 0..7
  const int ncol  = lane & 15;         // N within tile / M within A rows
  const int half  = lane >> 4;
  const int dir   = blockIdx.y;        // 0 = forward, 1 = backward
  const int bch   = blockIdx.x * 16;

  // h0 = c0 = 0
  for (int i = tid; i < 16 * H_; i += 256) {
    ((_Float16*)h_sh)[i] = (_Float16)0.0f;
    ((float*)c_sh)[i]    = 0.0f;
    ((float*)h32_sh)[i]  = 0.0f;
  }
  __syncthreads();

  // per-wave bias values for its 8 N-tiles (constant over time)
  float biasv[8];
#pragma unroll
  for (int q = 0; q < 8; ++q)
    biasv[q] = biasC[dir * G_ + (wave * 8 + q) * 16 + ncol];

  const unsigned int* whhD = whhF + (size_t)dir * WHH_FRAG_DW;
  const unsigned int* wihD = wihF + (size_t)dir * WIH_FRAG_DW;

  for (int s = 0; s < T_; ++s) {
    const int t = dir ? (T_ - 1 - s) : s;

    // ---- stage x_t tile [16 rows x 128] f32 -> f16 into LDS ----
    {
      int m  = tid >> 4;
      int i0 = (tid & 15) * 8;
      const float* xp = x + ((size_t)(bch + m) * T_ + t) * I_ + i0;
      float4 x0 = ((const float4*)xp)[0];
      float4 x1 = ((const float4*)xp)[1];
      _Float16* d = &xa_sh[m][i0];
      d[0] = (_Float16)x0.x; d[1] = (_Float16)x0.y;
      d[2] = (_Float16)x0.z; d[3] = (_Float16)x0.w;
      d[4] = (_Float16)x1.x; d[5] = (_Float16)x1.y;
      d[6] = (_Float16)x1.z; d[7] = (_Float16)x1.w;
    }
    __syncthreads();

    // ---- gates = bias + x_t @ Wih^T + h @ Whh^T  (WMMA f16 -> f32) ----
    v8f acc[8];
#pragma unroll
    for (int q = 0; q < 8; ++q)
#pragma unroll
      for (int c = 0; c < 8; ++c) acc[q][c] = biasv[q];

    // input projection: K = 128 (4 k-tiles of 32)
    for (int kt = 0; kt < KTX; ++kt) {
      Frag16 fa;
      const _Float16* ap = &xa_sh[ncol][kt * 32 + 16 * half];
      fa.q[0] = *(const uint4*)ap;
      fa.q[1] = *(const uint4*)(ap + 8);
#pragma unroll
      for (int q = 0; q < 8; ++q) {
        const int nt = wave * 8 + q;
        Frag16 fb;
        const uint4* gp = (const uint4*)(wihD + ((size_t)(nt * KTX + kt) * 32 + lane) * 8);
        fb.q[0] = gp[0]; fb.q[1] = gp[1];
        acc[q] = __builtin_amdgcn_wmma_f32_16x16x32_f16(
            false, fa.v, false, fb.v, (short)0, acc[q], false, false);
      }
    }
    // recurrent projection: K = 256 (8 k-tiles of 32)
    for (int kt = 0; kt < KTH; ++kt) {
      Frag16 fa;
      const _Float16* ap = &h_sh[ncol][kt * 32 + 16 * half];
      fa.q[0] = *(const uint4*)ap;
      fa.q[1] = *(const uint4*)(ap + 8);
#pragma unroll
      for (int q = 0; q < 8; ++q) {
        const int nt = wave * 8 + q;
        Frag16 fb;
        const uint4* gp = (const uint4*)(whhD + ((size_t)(nt * KTH + kt) * 32 + lane) * 8);
        fb.q[0] = gp[0]; fb.q[1] = gp[1];
        acc[q] = __builtin_amdgcn_wmma_f32_16x16x32_f16(
            false, fa.v, false, fb.v, (short)0, acc[q], false, false);
      }
    }

    // ---- spill accumulators: C layout lane=(n,half), vgpr v -> m = v + 8*half ----
#pragma unroll
    for (int q = 0; q < 8; ++q) {
      const int nt = wave * 8 + q;
      float* gpo = &gates_sh[nt * 16 + ncol][8 * half];
      float4 lo = {acc[q][0], acc[q][1], acc[q][2], acc[q][3]};
      float4 hi = {acc[q][4], acc[q][5], acc[q][6], acc[q][7]};
      ((float4*)gpo)[0] = lo;
      ((float4*)gpo)[1] = hi;
    }
    __syncthreads();

    // ---- elementwise LSTM cell update (torch gate order i,f,g,o) ----
    {
      const int m  = tid & 15;
      const int jb = tid >> 4;
#pragma unroll
      for (int jj = 0; jj < 16; ++jj) {
        const int j = jb * 16 + jj;
        float ig = gates_sh[j][m];
        float fg = gates_sh[H_ + j][m];
        float gg = gates_sh[2 * H_ + j][m];
        float og = gates_sh[3 * H_ + j][m];
        ig = 1.0f / (1.0f + __expf(-ig));
        fg = 1.0f / (1.0f + __expf(-fg));
        og = 1.0f / (1.0f + __expf(-og));
        gg = tanhf(gg);
        const float c2 = fg * c_sh[j][m] + ig * gg;
        const float h2 = og * tanhf(c2);
        c_sh[j][m]   = c2;
        h32_sh[j][m] = h2;
        h_sh[m][j]   = (_Float16)h2;
      }
    }
    __syncthreads();
  }

  // ---- write final hidden state (f32) ----
  for (int idx = tid; idx < 16 * H_; idx += 256) {
    const int j = idx & (H_ - 1);
    const int m = idx >> 8;
    hfinal[((size_t)dir * B_ + (bch + m)) * H_ + j] = h32_sh[j][m];
  }
}

// ---------------------------------------------------------------------------
// out[b] = b_fc + sum_j h_f[b][j]*Wfc[j] + sum_j h_b[b][j]*Wfc[256+j]
// ---------------------------------------------------------------------------
__global__ void fc_kernel(const float* __restrict__ hfinal,
                          const float* __restrict__ wfc,
                          const float* __restrict__ bfc,
                          float* __restrict__ out) {
  const int b = threadIdx.x;  // 256 threads, one per batch row
  float s = bfc[0];
  for (int j = 0; j < 2 * H_; ++j) {
    const float h = hfinal[(size_t)(j >> 8) * (B_ * H_) + (size_t)b * H_ + (j & (H_ - 1))];
    s += h * wfc[j];
  }
  out[b] = s;
}

extern "C" void kernel_launch(void* const* d_in, const int* in_sizes, int n_in,
                              void* d_out, int out_size, void* d_ws, size_t ws_size,
                              hipStream_t stream) {
  (void)in_sizes; (void)n_in; (void)out_size; (void)ws_size;
  const float* x     = (const float*)d_in[0];
  const float* Wih_f = (const float*)d_in[1];
  const float* Whh_f = (const float*)d_in[2];
  const float* bih_f = (const float*)d_in[3];
  const float* bhh_f = (const float*)d_in[4];
  const float* Wih_b = (const float*)d_in[5];
  const float* Whh_b = (const float*)d_in[6];
  const float* bih_b = (const float*)d_in[7];
  const float* bhh_b = (const float*)d_in[8];
  const float* W_fc  = (const float*)d_in[9];
  const float* b_fc  = (const float*)d_in[10];
  float* out = (float*)d_out;

  // workspace layout (~2.1 MB)
  unsigned int* whhF = (unsigned int*)d_ws;             // 2 * 131072 dwords
  unsigned int* wihF = whhF + 2 * WHH_FRAG_DW;          // 2 *  65536 dwords
  float* biasC  = (float*)(wihF + 2 * WIH_FRAG_DW);     // 2048 floats
  float* hfinal = biasC + 2 * G_;                       // 2*256*256 floats

  // pack weights into WMMA B-fragment order (f32 -> f16)
  conv_b_frag_kernel<<<(WHH_FRAG_DW + 255) / 256, 256, 0, stream>>>(
      Whh_f, whhF, H_, KTH);
  conv_b_frag_kernel<<<(WHH_FRAG_DW + 255) / 256, 256, 0, stream>>>(
      Whh_b, whhF + WHH_FRAG_DW, H_, KTH);
  conv_b_frag_kernel<<<(WIH_FRAG_DW + 255) / 256, 256, 0, stream>>>(
      Wih_f, wihF, I_, KTX);
  conv_b_frag_kernel<<<(WIH_FRAG_DW + 255) / 256, 256, 0, stream>>>(
      Wih_b, wihF + WIH_FRAG_DW, I_, KTX);
  bias_kernel<<<(2 * G_ + 255) / 256, 256, 0, stream>>>(
      bih_f, bhh_f, bih_b, bhh_b, biasC);

  // persistent recurrence: 16 batch-chunks x 2 directions
  dim3 grid(B_ / 16, 2);
  lstm_kernel<<<grid, 256, 0, stream>>>(x, whhF, wihF, biasC, hfinal);

  // final projection [B,1]
  fc_kernel<<<1, B_, 0, stream>>>(hfinal, W_fc, b_fc, out);
}
